// Corr1d_87119116632521
// MI455X (gfx1250) — matrix operand order repaired and checked by reading, not compiled, not run on hardware
//
#include <hip/hip_runtime.h>
#include <hip/hip_bf16.h>
#include <stdint.h>

// Problem constants (match reference setup_inputs)
#define B_   8
#define C_   128
#define H_   96
#define W_   320
#define CH   (H_ * W_)            // channel stride = 30720
#define SW   324                  // padded LDS row stride (words); 4*324%64=16, 8*324%64=32 -> conflict-free halves
#define NPAIR 16                  // channel pairs staged per pass (32 channels)
#define NPASS 4                   // 4 * 32 = 128 channels
#define NWAVES 10
#define BLOCK (NWAVES * 32)       // 320 threads, one per column w
#define SCALE (1.0f / (128.0f * 21.0f))

typedef __attribute__((ext_vector_type(16))) __bf16       v16bf;
typedef __attribute__((ext_vector_type(8)))  float        v8f;
typedef __attribute__((ext_vector_type(8)))  unsigned int v8u;

// round-to-nearest-even f32 -> bf16, packed pair (a in [15:0], b in [31:16])
__device__ inline uint32_t pack_bf16_rne(float a, float b) {
  uint32_t ua = __float_as_uint(a), ub = __float_as_uint(b);
  ua += 0x7FFFu + ((ua >> 16) & 1u);
  ub += 0x7FFFu + ((ub >> 16) & 1u);
  return (ua >> 16) | (ub & 0xFFFF0000u);
}

// A-matrix (16x32 bf16) fragment: lane = M row; ISA table:
//  lanes 0-15 : V0-3 K=0..7,  V4-7 K=16..23
//  lanes 16-31: V0-3 K=8..15, V4-7 K=24..31
// pair index p = K/2 stored as one u32 word at lds[p*SW + wcol]
__device__ inline v16bf load_frag_A(const uint32_t* t, int wcol, int hi) {
  v8u r;
#pragma unroll
  for (int j = 0; j < 8; ++j) {
    const int p = (j & 3) + ((j >> 2) << 3) + (hi ? 4 : 0);
    r[j] = t[p * SW + wcol];
  }
  return __builtin_bit_cast(v16bf, r);
}

// B-matrix (32x16 bf16) fragment: lane = N col; same K order as A:
// lanes 0-15 K=0..7,16..23; lanes 16-31 K=8..15,24..31
__device__ inline v16bf load_frag_B(const uint32_t* t, int wcol, int hi) {
  v8u r;
#pragma unroll
  for (int j = 0; j < 8; ++j) {
    const int p = (j & 3) + ((j >> 2) << 3) + (hi ? 4 : 0);
    r[j] = t[p * SW + wcol];
  }
  return __builtin_bit_cast(v16bf, r);
}

__global__ __launch_bounds__(BLOCK) void corr1d_wmma(const float* __restrict__ x,
                                                     const float* __restrict__ y,
                                                     float* __restrict__ out) {
  __shared__ uint32_t lds_x[NPAIR * SW];   // packed bf16 pairs: [pair p][w]
  __shared__ uint32_t lds_y[NPAIR * SW];

  const int tid  = threadIdx.x;            // 0..319 == column w for staging
  const int lane = tid & 31;
  const int wave = tid >> 5;               // 0..9
  const int hi   = lane >> 4;
  const int n    = lane & 15;              // column within tile (and A row M)
  const int bh   = blockIdx.x;             // b*H + h
  const int b    = bh / H_;
  const int h    = bh % H_;
  const size_t base = (size_t)b * C_ * CH + (size_t)h * W_;

  // each wave owns two diagonal 16-col positions: q and q+10
  const int c0s[2] = { wave * 16, (wave + NWAVES) * 16 };

  v8f zero = {};
  v8f acc[2][4];                           // [pos][ diagXY, subXY, diagYX, superYX ]
#pragma unroll
  for (int p = 0; p < 2; ++p)
#pragma unroll
    for (int t = 0; t < 4; ++t) acc[p][t] = zero;

#pragma unroll 1
  for (int pass = 0; pass < NPASS; ++pass) {
    __syncthreads();                       // previous pass done reading LDS
    const int cbase = pass * (2 * NPAIR);
#pragma unroll
    for (int it = 0; it < NPAIR; ++it) {   // stage 32 channels, coalesced over w
      const size_t off = base + (size_t)(cbase + 2 * it) * CH + tid;
      lds_x[it * SW + tid] = pack_bf16_rne(x[off], x[off + CH]);
      lds_y[it * SW + tid] = pack_bf16_rne(y[off], y[off + CH]);
    }
    __syncthreads();                       // staged data visible

#pragma unroll
    for (int p = 0; p < 2; ++p) {
      const int c0 = c0s[p];
      const int wc = c0 + n;
      // G = X^T Y : diagonal tile + sub-diagonal tile (rows c0+16..c0+31)
      v16bf yB  = load_frag_B(lds_y, wc, hi);
      v16bf xA0 = load_frag_A(lds_x, wc, hi);
      acc[p][0] = __builtin_amdgcn_wmma_f32_16x16x32_bf16(
          false, xA0, false, yB, (short)0, acc[p][0], false, false);
      if (c0 + 16 < W_) {                  // wave-uniform; skip OOB rows >= W
        v16bf xA1 = load_frag_A(lds_x, wc + 16, hi);
        acc[p][1] = __builtin_amdgcn_wmma_f32_16x16x32_bf16(
            false, xA1, false, yB, (short)0, acc[p][1], false, false);
      }
      // G~ = Y^T X : diagonal tile + super-diagonal tile (rows c0-16..c0-1)
      v16bf xB  = load_frag_B(lds_x, wc, hi);
      v16bf yA0 = load_frag_A(lds_y, wc, hi);
      acc[p][2] = __builtin_amdgcn_wmma_f32_16x16x32_bf16(
          false, yA0, false, xB, (short)0, acc[p][2], false, false);
      if (c0 >= 16) {                      // wave-uniform; skip OOB rows < 0
        v16bf yAm = load_frag_A(lds_y, wc - 16, hi);
        acc[p][3] = __builtin_amdgcn_wmma_f32_16x16x32_bf16(
            false, yAm, false, xB, (short)0, acc[p][3], false, false);
      }
    }
  }

  // Band reduction. C/D layout: VGPR r, lanes 0-15 -> M=r, lanes 16-31 -> M=8+r,
  // lane%16 = N. Each lane sums its column's band entries, then combine halves.
#pragma unroll
  for (int p = 0; p < 2; ++p) {
    const int c0 = c0s[p];
    const int mb = hi ? 8 : 0;
    float s = 0.0f;
#pragma unroll
    for (int r = 0; r < 8; ++r) {
      const int d = mb + r - n;            // row M minus col N
      if (d >= 0 && d <= 10)   s += acc[p][0][r];  // G[w+d, w], d=0..10 (diag)
      if (d <= -6)             s += acc[p][1][r];  // G[w+d, w], d=1..10 (sub-diag)
      if (d >= -10 && d <= -1) s += acc[p][2][r];  // G[w, w-d], d=1..10 (diag of G~)
      if (d >= 6)              s += acc[p][3][r];  // G[w, w-d], d=1..10 (super-diag of G~)
    }
    s += __shfl_xor(s, 16, 32);            // combine M-halves of the column
    if (lane < 16) out[(size_t)bh * W_ + c0 + n] = s * SCALE;
  }
}

extern "C" void kernel_launch(void* const* d_in, const int* in_sizes, int n_in,
                              void* d_out, int out_size, void* d_ws, size_t ws_size,
                              hipStream_t stream) {
  (void)in_sizes; (void)n_in; (void)d_ws; (void)ws_size; (void)out_size;
  const float* x = (const float*)d_in[0];
  const float* y = (const float*)d_in[1];
  float* out = (float*)d_out;
  dim3 grid(B_ * H_);                      // one workgroup per (b, h) row
  dim3 block(BLOCK);                       // 320 threads = 10 wave32
  corr1d_wmma<<<grid, block, 0, stream>>>(x, y, out);
}
